// MultiHeadAttention_20650202759187
// MI455X (gfx1250) — matrix operand order repaired and checked by reading, not compiled, not run on hardware
//
#include <hip/hip_runtime.h>
#include <hip/hip_bf16.h>

// ---------------------------------------------------------------------------
// MI455X (gfx1250, wave32) multi-head causal attention, f16 WMMA / f32 acc.
// GEMMs use GLOBAL_LOAD_ASYNC_TO_LDS_B128 double-buffered software pipeline.
// ---------------------------------------------------------------------------

typedef __attribute__((ext_vector_type(16))) _Float16 v16h;
typedef __attribute__((ext_vector_type(8)))  _Float16 v8h;
typedef __attribute__((ext_vector_type(4)))  _Float16 v4h;
typedef __attribute__((ext_vector_type(8)))  float    v8f;

#define WMMA_F16(a, b, c) \
  __builtin_amdgcn_wmma_f32_16x16x32_f16(false, (a), false, (b), (short)0, (c), false, false)

static __device__ __forceinline__ v16h cat16(v8h lo, v8h hi) {
  return __builtin_shufflevector(lo, hi, 0, 1, 2, 3, 4, 5, 6, 7, 8, 9, 10, 11, 12, 13, 14, 15);
}

// Async global->LDS copy (CDNA5, tracked by ASYNCcnt). Each active lane copies
// 16 bytes from its global address to its LDS byte offset (flat-shared low 32b).
static __device__ __forceinline__ void async_ld_b128(unsigned lds, const _Float16* g) {
  asm volatile("global_load_async_to_lds_b128 %0, %1, off"
               :
               : "v"(lds), "v"((unsigned long long)(uintptr_t)g)
               : "memory");
}
static __device__ __forceinline__ void wait_async0() {
  asm volatile("s_wait_asynccnt 0x0" ::: "memory");
}
static __device__ __forceinline__ unsigned lds_off(const void* p) {
  return (unsigned)(unsigned long long)(uintptr_t)p;   // low 32 bits give the LDS offset
}

// ------------------------------ cast kernels -------------------------------

__global__ void cast_f32_f16_kernel(const float* __restrict__ in,
                                    _Float16* __restrict__ out, int n4) {
  int i = blockIdx.x * blockDim.x + threadIdx.x;
  if (i >= n4) return;
  float4 f = ((const float4*)in)[i];
  v4h h;
  h[0] = (_Float16)f.x; h[1] = (_Float16)f.y;
  h[2] = (_Float16)f.z; h[3] = (_Float16)f.w;
  ((v4h*)out)[i] = h;
}

// W [H=16, E=1024, d=64] fp32  ->  W' [E, H*d=1024] f16 ; out idx = e*1024 + h*64 + dd
__global__ void cast_w_hed_to_ehd(const float* __restrict__ in,
                                  _Float16* __restrict__ out) {
  int idx = blockIdx.x * 256 + threadIdx.x;       // over 1M elements
  int e  = idx >> 10;
  int c  = idx & 1023;
  int h  = c >> 6;
  int dd = c & 63;
  out[idx] = (_Float16)in[(h << 16) + (e << 6) + dd];   // h*E*d + e*d + dd
}

// ------------------------------ tiled GEMM ---------------------------------
// C[M,N] = (A[M,K] @ B[K,N] + bias[N]) * scale ; A,B f16, acc f32.
// Block tile 256x64, BK=32, 256 threads = 8 waves; wave w owns rows
// [w*32, w*32+32) = two 16-row WMMA strips. A tile staged by async
// global->LDS copies, double-buffered; B tile staged transposed via VGPRs.

template <bool OUT_F32>
__global__ __launch_bounds__(256) void gemm_bias_kernel(
    const _Float16* __restrict__ A, const _Float16* __restrict__ B,
    const float* __restrict__ bias, void* __restrict__ C,
    int M, int N, int K, float scale) {
  __shared__ __align__(32) _Float16 ldsA[2][256 * 32];
  __shared__ __align__(32) _Float16 ldsB[2][64 * 32];   // transposed: [n][k]

  const int tid  = threadIdx.x;
  const int w    = tid >> 5;
  const int lane = tid & 31;
  const int half = lane >> 4;
  const int ml   = lane & 15;
  const int m0   = blockIdx.x * 256;
  const int n0   = blockIdx.y * 64;

  v8f acc[2][4] = {};

  const int bk = tid >> 3, bn = (tid & 7) * 8;   // B staging: 8 halfs/thread
  const unsigned ldsA0 = lds_off(&ldsA[0][0]);

  // thread tid stages A row (m0+tid): 32 halfs = 4 async b128 copies
  auto issueA = [&](int kk, int bb) {
    const _Float16* ga = A + (size_t)(m0 + tid) * K + kk;
    unsigned dst = ldsA0 + (unsigned)(bb * (256 * 32) + tid * 32) * 2u;
#pragma unroll
    for (int p = 0; p < 4; ++p) async_ld_b128(dst + p * 16u, ga + p * 8);
  };
  auto stageB = [&](int kk, int bb) {
    v8h bvv = *(const v8h*)(B + (size_t)(kk + bk) * N + n0 + bn);
    _Float16* d = &ldsB[bb][0];
#pragma unroll
    for (int i = 0; i < 8; ++i) d[(bn + i) * 32 + bk] = bvv[i];
  };

  issueA(0, 0);
  stageB(0, 0);
  wait_async0();
  __syncthreads();

  const int NIT = K >> 5;
  for (int it = 0; it < NIT; ++it) {
    const int buf = it & 1;
    if (it + 1 < NIT) {           // prefetch next tile while computing this one
      issueA((it + 1) << 5, buf ^ 1);
      stageB((it + 1) << 5, buf ^ 1);
    }

    v16h afr[2];
#pragma unroll
    for (int ss = 0; ss < 2; ++ss) {
      const _Float16* ar = &ldsA[buf][(w * 32 + ss * 16 + ml) * 32];
      afr[ss] = cat16(*(const v8h*)(ar + half * 16),
                      *(const v8h*)(ar + half * 16 + 8));
    }
#pragma unroll
    for (int nn = 0; nn < 4; ++nn) {
      v16h b = *(const v16h*)(&ldsB[buf][(nn * 16 + ml) * 32 + half * 16]);
      acc[0][nn] = WMMA_F16(afr[0], b, acc[0][nn]);
      acc[1][nn] = WMMA_F16(afr[1], b, acc[1][nn]);
    }

    wait_async0();                // own async copies of next tile complete
    __syncthreads();              // everyone's copies done / readers drained
  }

#pragma unroll
  for (int ss = 0; ss < 2; ++ss)
#pragma unroll
    for (int nn = 0; nn < 4; ++nn) {
      const int col = n0 + nn * 16 + ml;
      const float bv = bias[col];
#pragma unroll
      for (int j = 0; j < 8; ++j) {
        const int row = m0 + w * 32 + ss * 16 + half * 8 + j;
        float v = (acc[ss][nn][j] + bv) * scale;
        if (OUT_F32) ((float*)C)[(size_t)row * N + col] = v;
        else ((_Float16*)C)[(size_t)row * N + col] = (_Float16)v;
      }
    }
}

// --------------------------- flash attention -------------------------------
// Q/K/V f16 laid out [B*T, H*64] (col = h*64 + dd); q pre-scaled by 1/sqrt(d).
// Block: 128 query rows for one (b, h); 8 waves, 16 rows each. Causal.

__global__ __launch_bounds__(256) void attn_fwd_kernel(
    const _Float16* __restrict__ Q, const _Float16* __restrict__ Kmat,
    const _Float16* __restrict__ V, _Float16* __restrict__ O, int T) {
  __shared__ __align__(32) _Float16 ldsV[64 * 128];      // transposed: [d][key]
  __shared__ __align__(32) _Float16 ldsP[8 * 16 * 128];  // per-wave P: [q][key]

  const int tid  = threadIdx.x;
  const int w    = tid >> 5;
  const int lane = tid & 31;
  const int half = lane >> 4;
  const int ml   = lane & 15;
  const int mt = blockIdx.x, hh = blockIdx.y, b = blockIdx.z;

  const size_t base = (size_t)b * T * 1024 + (size_t)hh * 64;

  // Q fragments (A layout): lane m = w*16+ml, K slots over d = 64 (2 chunks of 32)
  const int qrow_g = mt * 128 + w * 16 + ml;
  const _Float16* qp = Q + base + (size_t)qrow_g * 1024;
  v16h qf[2];
#pragma unroll
  for (int c = 0; c < 2; ++c)
    qf[c] = cat16(*(const v8h*)(qp + c * 32 + half * 16),
                  *(const v8h*)(qp + c * 32 + half * 16 + 8));

  v8f of[4] = {};
  float mrow[8], lrow[8];
#pragma unroll
  for (int j = 0; j < 8; ++j) { mrow[j] = -3.0e38f; lrow[j] = 0.0f; }

  const int vkey = tid >> 1, vdh = (tid & 1) * 32;   // V staging indices

  for (int kt = 0; kt <= mt; ++kt) {
    // ---- stage V tile (128 keys x 64 d), transposed into LDS ----
    {
      const _Float16* vp = V + base + (size_t)(kt * 128 + vkey) * 1024 + vdh;
      v8h vv[4];
#pragma unroll
      for (int q = 0; q < 4; ++q) vv[q] = *(const v8h*)(vp + q * 8);
#pragma unroll
      for (int q = 0; q < 4; ++q)
#pragma unroll
        for (int i = 0; i < 8; ++i)
          ldsV[(vdh + q * 8 + i) * 128 + vkey] = vv[q][i];
    }
    __syncthreads();

    // ---- S = Q @ K^T (8 key sub-tiles of 16); K loaded direct (B layout) ----
    v8f s[8];
#pragma unroll
    for (int ks = 0; ks < 8; ++ks) {
      const int key_g = kt * 128 + ks * 16 + ml;
      const _Float16* kp = Kmat + base + (size_t)key_g * 1024;
      v8f sa = {};
#pragma unroll
      for (int c = 0; c < 2; ++c) {
        v16h bfr = *(const v16h*)(kp + c * 32 + half * 16);
        sa = WMMA_F16(qf[c], bfr, sa);
      }
      s[ks] = sa;
    }

    // ---- causal mask on diagonal tile ----
    if (kt == mt) {
#pragma unroll
      for (int ks = 0; ks < 8; ++ks)
#pragma unroll
        for (int j = 0; j < 8; ++j)
          if (ks * 16 + ml > w * 16 + half * 8 + j) s[ks][j] = -3.0e38f;
    }

    // ---- online softmax (row stats replicated across the 16-lane half) ----
    float mnew[8], alpha[8];
#pragma unroll
    for (int j = 0; j < 8; ++j) {
      float t = s[0][j];
#pragma unroll
      for (int ks = 1; ks < 8; ++ks) t = fmaxf(t, s[ks][j]);
#pragma unroll
      for (int msk = 1; msk < 16; msk <<= 1) t = fmaxf(t, __shfl_xor(t, msk, 32));
      mnew[j]  = fmaxf(mrow[j], t);
      alpha[j] = __expf(mrow[j] - mnew[j]);
      mrow[j]  = mnew[j];
    }
#pragma unroll
    for (int ks = 0; ks < 8; ++ks)
#pragma unroll
      for (int j = 0; j < 8; ++j) s[ks][j] = __expf(s[ks][j] - mnew[j]);
#pragma unroll
    for (int j = 0; j < 8; ++j) {
      float r = 0.0f;
#pragma unroll
      for (int ks = 0; ks < 8; ++ks) r += s[ks][j];
#pragma unroll
      for (int msk = 1; msk < 16; msk <<= 1) r += __shfl_xor(r, msk, 32);
      lrow[j] = lrow[j] * alpha[j] + r;
    }
#pragma unroll
    for (int nn = 0; nn < 4; ++nn)
#pragma unroll
      for (int j = 0; j < 8; ++j) of[nn][j] *= alpha[j];

    // ---- P -> per-wave LDS (C layout -> [q][key]) ----
    _Float16* pw = ldsP + w * 2048;
#pragma unroll
    for (int ks = 0; ks < 8; ++ks)
#pragma unroll
      for (int j = 0; j < 8; ++j)
        pw[(half * 8 + j) * 128 + ks * 16 + ml] = (_Float16)s[ks][j];

    // ---- O += P @ V  (A frags from ldsP, B frags from transposed ldsV) ----
#pragma unroll
    for (int c = 0; c < 4; ++c) {
      const _Float16* pr = ldsP + w * 2048 + ml * 128 + c * 32;
      v16h a = cat16(*(const v8h*)(pr + half * 16),
                     *(const v8h*)(pr + half * 16 + 8));
#pragma unroll
      for (int nn = 0; nn < 4; ++nn) {
        v16h bfr = *(const v16h*)(ldsV + (nn * 16 + ml) * 128 + c * 32 + half * 16);
        of[nn] = WMMA_F16(a, bfr, of[nn]);
      }
    }
    __syncthreads();
  }

  // ---- epilogue: normalize, store f16 [B*T, H*64] ----
#pragma unroll
  for (int nn = 0; nn < 4; ++nn)
#pragma unroll
    for (int j = 0; j < 8; ++j) {
      const int trow = mt * 128 + w * 16 + half * 8 + j;
      O[base + (size_t)trow * 1024 + nn * 16 + ml] = (_Float16)(of[nn][j] / lrow[j]);
    }
}

// ------------------------------- launcher ----------------------------------

extern "C" void kernel_launch(void* const* d_in, const int* in_sizes, int n_in,
                              void* d_out, int out_size, void* d_ws, size_t ws_size,
                              hipStream_t stream) {
  const float* x  = (const float*)d_in[0];
  const float* Wq = (const float*)d_in[1];
  const float* Wk = (const float*)d_in[2];
  const float* Wv = (const float*)d_in[3];
  const float* bq = (const float*)d_in[4];
  const float* bk = (const float*)d_in[5];
  const float* bv = (const float*)d_in[6];
  const float* Wp = (const float*)d_in[7];
  const float* bp = (const float*)d_in[8];

  const int B = 4, T = 2048, E = 1024;
  const int BT = B * T;  // 8192

  // Workspace carve-out (~88 MB total, 256B aligned slices).
  char* ws = (char*)d_ws;
  size_t off = 0;
  auto carve = [&](size_t bytes) -> _Float16* {
    char* p = ws + off;
    off += (bytes + 255) & ~(size_t)255;
    return (_Float16*)p;
  };
  _Float16* xh  = carve((size_t)BT * E * 2);  // x f16
  _Float16* wqh = carve((size_t)E * E * 2);   // Wq [E, H*d] f16
  _Float16* wkh = carve((size_t)E * E * 2);
  _Float16* wvh = carve((size_t)E * E * 2);
  _Float16* wph = carve((size_t)E * E * 2);
  _Float16* qh  = carve((size_t)BT * E * 2);  // q [B,T,H,d] f16 (pre-scaled)
  _Float16* kh  = carve((size_t)BT * E * 2);
  _Float16* vh  = carve((size_t)BT * E * 2);
  _Float16* att = carve((size_t)BT * E * 2);  // attention output f16
  (void)ws_size; (void)in_sizes; (void)n_in; (void)out_size;

  // 1) casts / layout transforms
  cast_f32_f16_kernel<<<(BT * E / 4) / 256, 256, 0, stream>>>(x, xh, BT * E / 4);
  cast_w_hed_to_ehd<<<(E * E) / 256, 256, 0, stream>>>(Wq, wqh);
  cast_w_hed_to_ehd<<<(E * E) / 256, 256, 0, stream>>>(Wk, wkh);
  cast_w_hed_to_ehd<<<(E * E) / 256, 256, 0, stream>>>(Wv, wvh);
  cast_f32_f16_kernel<<<(E * E / 4) / 256, 256, 0, stream>>>(Wp, wph, E * E / 4);

  // 2) fused all-head QKV projections: [8192,1024] @ [1024,1024] (+bias)
  dim3 gg(BT / 256, E / 64);  // 32 x 16
  gemm_bias_kernel<false><<<gg, 256, 0, stream>>>(xh, wqh, bq, qh, BT, E, E, 0.125f);
  gemm_bias_kernel<false><<<gg, 256, 0, stream>>>(xh, wkh, bk, kh, BT, E, E, 1.0f);
  gemm_bias_kernel<false><<<gg, 256, 0, stream>>>(xh, wvh, bv, vh, BT, E, E, 1.0f);

  // 3) causal flash attention
  attn_fwd_kernel<<<dim3(T / 128, 16, B), 256, 0, stream>>>(qh, kh, vh, att, T);

  // 4) output projection -> fp32 d_out
  gemm_bias_kernel<true><<<gg, 256, 0, stream>>>(att, wph, bp, d_out, BT, E, E, 1.0f);
}
